// Attention_1305670058248
// MI455X (gfx1250) — matrix operand order repaired and checked by reading, not compiled, not run on hardware
//
#include <hip/hip_runtime.h>

typedef __attribute__((ext_vector_type(16))) _Float16 v16h;
typedef __attribute__((ext_vector_type(8)))  _Float16 v8h;
typedef __attribute__((ext_vector_type(8)))  float    v8f;

// exact vector type the async-LDS builtin expects (from the compiler diagnostic)
typedef int i32x4 __attribute__((vector_size(4 * sizeof(int))));
typedef __attribute__((address_space(1))) i32x4 g_i32x4;   // global
typedef __attribute__((address_space(3))) i32x4 l_i32x4;   // LDS

// Problem constants
constexpr int S    = 2048;
constexpr int D    = 2048;
constexpr int H    = 32;
constexpr int KV   = 8;
constexpr int HD   = 64;
constexpr int NREP = H / KV;          // 4
constexpr int HHD  = H * HD;          // 2048
constexpr int KVHD = KV * HD;         // 512

// ---------------------------------------------------------------------------
// CDNA5 async global->LDS copy (16B per lane) + ASYNCcnt wait.
// ---------------------------------------------------------------------------
#if defined(__has_builtin)
#  if __has_builtin(__builtin_amdgcn_global_load_async_to_lds_b128)
#    define HAVE_ASYNC_BUILTIN 1
#  endif
#  if __has_builtin(__builtin_amdgcn_s_wait_asynccnt)
#    define HAVE_WAIT_BUILTIN 1
#  endif
#endif
#ifndef HAVE_ASYNC_BUILTIN
#define HAVE_ASYNC_BUILTIN 0
#endif
#ifndef HAVE_WAIT_BUILTIN
#define HAVE_WAIT_BUILTIN 0
#endif

__device__ __forceinline__ void cp16(void* l, const void* g) {
#if HAVE_ASYNC_BUILTIN
    __builtin_amdgcn_global_load_async_to_lds_b128(
        (g_i32x4*)g, (l_i32x4*)l, 0, 0);
#else
    unsigned laddr = (unsigned)(unsigned long long)(
        (__attribute__((address_space(3))) void*)l);
    asm volatile("global_load_async_to_lds_b128 %0, %1, off"
                 :: "v"(laddr), "v"(g) : "memory");
#endif
}

template <int N>
__device__ __forceinline__ void cp_wait() {
#if HAVE_WAIT_BUILTIN
    __builtin_amdgcn_s_wait_asynccnt(N);
#else
    asm volatile("s_wait_asynccnt %0" :: "i"(N) : "memory");
#endif
}

// ---------------------------------------------------------------------------
// f32 -> f16 convert (plain)
// ---------------------------------------------------------------------------
__global__ void cvt_f16_kernel(const float* __restrict__ src,
                               _Float16* __restrict__ dst, int n) {
    int i = blockIdx.x * blockDim.x + threadIdx.x;
    if (i < n) dst[i] = (_Float16)src[i];
}

// ---------------------------------------------------------------------------
// f32 -> f16 convert with transpose: dst(Ndim x Kdim) = src(Kdim x Ndim)^T
// ---------------------------------------------------------------------------
__global__ __launch_bounds__(256)
void cvt_transpose_kernel(const float* __restrict__ src,
                          _Float16* __restrict__ dst, int Kdim, int Ndim) {
    __shared__ float tile[32][33];
    int bn = blockIdx.x * 32;            // along Ndim
    int bk = blockIdx.y * 32;            // along Kdim
    int tx = threadIdx.x & 31;
    int ty = threadIdx.x >> 5;           // 0..7
#pragma unroll
    for (int i = 0; i < 32; i += 8)
        tile[ty + i][tx] = src[(size_t)(bk + ty + i) * Ndim + bn + tx];
    __syncthreads();
#pragma unroll
    for (int i = 0; i < 32; i += 8)
        dst[(size_t)(bn + ty + i) * Kdim + bk + tx] = (_Float16)tile[tx][ty + i];
}

// ---------------------------------------------------------------------------
// RoPE: src f32 (S, nheads*64) -> dst f16 (roped)
// ---------------------------------------------------------------------------
__global__ void rope_kernel(const float* __restrict__ src,
                            const float* __restrict__ cosb,
                            const float* __restrict__ sinb,
                            _Float16* __restrict__ dst, int nheads) {
    int idx = blockIdx.x * blockDim.x + threadIdx.x;   // over S*nheads*32
    int total = S * nheads * 32;
    if (idx >= total) return;
    int i = idx & 31;
    int h = (idx >> 5) % nheads;
    int s = idx / (32 * nheads);
    size_t row = (size_t)s * (nheads * 64) + h * 64;
    float x1 = src[row + i];
    float x2 = src[row + 32 + i];
    float c  = cosb[s * 32 + i];
    float sn = sinb[s * 32 + i];
    dst[row + i]      = (_Float16)(x1 * c - x2 * sn);
    dst[row + 32 + i] = (_Float16)(x2 * c + x1 * sn);
}

// ---------------------------------------------------------------------------
// Tiled WMMA GEMM: C(f32, MxN) = A(f16, MxK row-major) * Bt(f16, NxK row-major)
// BM=128 BN=128 BK=32, 256 threads (8 waves), wave tile 32x64.
// Double-buffered LDS fed by async global->LDS copies.
// ---------------------------------------------------------------------------
constexpr int BM = 128, BN = 128, BK = 32, GPAD = 8;

__global__ __launch_bounds__(256)
void gemm_f16_kernel(const _Float16* __restrict__ A,
                     const _Float16* __restrict__ Bt,
                     float* __restrict__ C, int M, int N, int K) {
    __shared__ _Float16 As[2][BM][BK + GPAD];   // (m, k)
    __shared__ _Float16 Bs[2][BN][BK + GPAD];   // (n, k)

    const int tid  = threadIdx.x;
    const int m0   = blockIdx.x * BM;
    const int n0   = blockIdx.y * BN;
    const int w    = tid >> 5;
    const int wm   = w & 3;                  // 4 waves along M (32 rows each)
    const int wn   = w >> 2;                 // 2 waves along N (64 cols each)
    const int lane = tid & 31;
    const int lh   = lane >> 4;
    const int lr   = lane & 15;

    // straight-line copy coords: chunks c = tid and tid+256
    const int cr  = tid >> 2;                // row 0..63 (t=0), +64 for t=1
    const int ckc = (tid & 3) << 3;          // k offset 0/8/16/24
    const _Float16* gA0 = A  + (size_t)(m0 + cr)      * K + ckc;
    const _Float16* gA1 = A  + (size_t)(m0 + cr + 64) * K + ckc;
    const _Float16* gB0 = Bt + (size_t)(n0 + cr)      * K + ckc;
    const _Float16* gB1 = Bt + (size_t)(n0 + cr + 64) * K + ckc;

    v8f acc[2][4] = {};

    // prologue: stage 0
    cp16(&As[0][cr][ckc],      gA0);
    cp16(&As[0][cr + 64][ckc], gA1);
    cp16(&Bs[0][cr][ckc],      gB0);
    cp16(&Bs[0][cr + 64][ckc], gB1);

    int p = 0;
    for (int k0 = 0; k0 < K; k0 += BK, p ^= 1) {
        if (k0 + BK < K) {
            int kn = k0 + BK;
            cp16(&As[p ^ 1][cr][ckc],      gA0 + kn);
            cp16(&As[p ^ 1][cr + 64][ckc], gA1 + kn);
            cp16(&Bs[p ^ 1][cr][ckc],      gB0 + kn);
            cp16(&Bs[p ^ 1][cr + 64][ckc], gB1 + kn);
            cp_wait<4>();
        } else {
            cp_wait<0>();
        }
        __syncthreads();

        v16h af[2], bf[4];
#pragma unroll
        for (int i = 0; i < 2; ++i) {
            int row = wm * 32 + i * 16 + lr;
            union { v16h v; v8h h[2]; } u;
            u.h[0] = *(const v8h*)&As[p][row][lh * 8];
            u.h[1] = *(const v8h*)&As[p][row][lh * 8 + 16];
            af[i] = u.v;
        }
#pragma unroll
        for (int j = 0; j < 4; ++j) {
            int col = wn * 64 + j * 16 + lr;
            union { v16h v; v8h h[2]; } u;
            u.h[0] = *(const v8h*)&Bs[p][col][lh * 8];
            u.h[1] = *(const v8h*)&Bs[p][col][lh * 8 + 16];
            bf[j] = u.v;
        }
#pragma unroll
        for (int i = 0; i < 2; ++i)
#pragma unroll
            for (int j = 0; j < 4; ++j)
                acc[i][j] = __builtin_amdgcn_wmma_f32_16x16x32_f16(
                    false, af[i], false, bf[j], (short)0, acc[i][j],
                    false, false);
        __syncthreads();
    }

#pragma unroll
    for (int i = 0; i < 2; ++i)
#pragma unroll
        for (int j = 0; j < 4; ++j)
#pragma unroll
            for (int r = 0; r < 8; ++r) {
                int row = m0 + wm * 32 + i * 16 + r + lh * 8;
                int col = n0 + wn * 64 + j * 16 + lr;
                C[(size_t)row * N + col] = acc[i][j][r];
            }
}

// ---------------------------------------------------------------------------
// Flash attention (causal, GQA): grid (H, S/64), 128 threads = 4 waves.
// V is pre-transposed globally: vtb (KV*HD, S). Double-buffered K/V tiles.
// ---------------------------------------------------------------------------
__global__ __launch_bounds__(128)
void attn_kernel(const _Float16* __restrict__ qh,
                 const _Float16* __restrict__ kh,
                 const _Float16* __restrict__ vtb,
                 _Float16* __restrict__ oh) {
    const int h    = blockIdx.x;
    const int qt   = blockIdx.y;
    const int q0   = qt * 64;
    const int kvh  = h / NREP;
    const int tid  = threadIdx.x;
    const int wq_  = tid >> 5;
    const int lane = tid & 31;
    const int lh   = lane >> 4;
    const int lr   = lane & 15;

    __shared__ _Float16 Qs[64][72];
    __shared__ _Float16 Ks[2][64][72];       // (k, d)
    __shared__ _Float16 Vt[2][64][72];       // (d, k)
    __shared__ _Float16 Ps[4][16][72];       // per-wave P strip

    // straight-line copy coords: chunks c = tid + 128t
    const int crow = tid >> 3;               // 0..15 (+16t)
    const int cdc  = (tid & 7) << 3;

    // Q tile (async, drained together with the first K/V stage)
#pragma unroll
    for (int t = 0; t < 4; ++t)
        cp16(&Qs[crow + 16 * t][cdc],
             qh + (size_t)(q0 + crow + 16 * t) * HHD + h * HD + cdc);

    const _Float16* gK = kh  + (size_t)crow * KVHD + kvh * HD + cdc;
    const _Float16* gV = vtb + (size_t)(kvh * HD + crow) * S + cdc;

    // prologue: K/V stage 0
#pragma unroll
    for (int t = 0; t < 4; ++t)
        cp16(&Ks[0][crow + 16 * t][cdc], gK + (size_t)(16 * t) * KVHD);
#pragma unroll
    for (int t = 0; t < 4; ++t)
        cp16(&Vt[0][crow + 16 * t][cdc], gV + (size_t)(16 * t) * S);

    v8f o[4] = {};
    float m_run[8], l_run[8];
#pragma unroll
    for (int r = 0; r < 8; ++r) { m_run[r] = -1e30f; l_run[r] = 0.f; }

    const float scale = 0.125f;              // HD^-0.5

    int p = 0;
    for (int kt = 0; kt <= qt; ++kt, p ^= 1) {
        const int k0 = kt * 64;
        if (kt < qt) {
            const int kn0 = k0 + 64;
#pragma unroll
            for (int t = 0; t < 4; ++t)
                cp16(&Ks[p ^ 1][crow + 16 * t][cdc],
                     gK + (size_t)(kn0 + 16 * t) * KVHD);
#pragma unroll
            for (int t = 0; t < 4; ++t)
                cp16(&Vt[p ^ 1][crow + 16 * t][cdc],
                     gV + (size_t)(16 * t) * S + kn0);
            cp_wait<8>();
        } else {
            cp_wait<0>();
        }
        __syncthreads();

        // ---- scores S = Q K^T : wave's 16 rows x 64 keys
        v8f sc[4] = {};
#pragma unroll
        for (int ks = 0; ks < 2; ++ks) {
            union { v16h v; v8h h[2]; } ua;
            int qrow = wq_ * 16 + lr;
            ua.h[0] = *(const v8h*)&Qs[qrow][ks * 32 + lh * 8];
            ua.h[1] = *(const v8h*)&Qs[qrow][ks * 32 + lh * 8 + 16];
#pragma unroll
            for (int j = 0; j < 4; ++j) {
                union { v16h v; v8h h[2]; } ub;
                ub.h[0] = *(const v8h*)&Ks[p][j * 16 + lr][ks * 32 + lh * 8];
                ub.h[1] = *(const v8h*)&Ks[p][j * 16 + lr][ks * 32 + lh * 8 + 16];
                sc[j] = __builtin_amdgcn_wmma_f32_16x16x32_f16(
                    false, ua.v, false, ub.v, (short)0, sc[j], false, false);
            }
        }

        // ---- online softmax update (per query row)
        const bool diag = (kt == qt);
#pragma unroll
        for (int r = 0; r < 8; ++r) {
            int grow = q0 + wq_ * 16 + r + lh * 8;
            float mx = -1e30f;
#pragma unroll
            for (int j = 0; j < 4; ++j) {
                float s = sc[j][r] * scale;
                if (diag && (k0 + j * 16 + lr) > grow) s = -1e30f;
                sc[j][r] = s;
                mx = fmaxf(mx, s);
            }
#pragma unroll
            for (int off = 1; off <= 8; off <<= 1)
                mx = fmaxf(mx, __shfl_xor(mx, off, 32));
            float mnew  = fmaxf(m_run[r], mx);
            float alpha = __expf(m_run[r] - mnew);
            m_run[r] = mnew;
            float psum = 0.f;
#pragma unroll
            for (int j = 0; j < 4; ++j) {
                float pv = __expf(sc[j][r] - mnew);
                sc[j][r] = pv;
                psum += pv;
            }
#pragma unroll
            for (int off = 1; off <= 8; off <<= 1)
                psum += __shfl_xor(psum, off, 32);
            l_run[r] = l_run[r] * alpha + psum;
#pragma unroll
            for (int j = 0; j < 4; ++j) o[j][r] *= alpha;
#pragma unroll
            for (int j = 0; j < 4; ++j)
                Ps[wq_][r + lh * 8][j * 16 + lr] = (_Float16)sc[j][r];
        }

        // ---- O += P V
#pragma unroll
        for (int ks = 0; ks < 2; ++ks) {
            union { v16h v; v8h h[2]; } ua;
            ua.h[0] = *(const v8h*)&Ps[wq_][lr][ks * 32 + lh * 8];
            ua.h[1] = *(const v8h*)&Ps[wq_][lr][ks * 32 + lh * 8 + 16];
#pragma unroll
            for (int j = 0; j < 4; ++j) {
                union { v16h v; v8h h[2]; } ub;
                ub.h[0] = *(const v8h*)&Vt[p][j * 16 + lr][ks * 32 + lh * 8];
                ub.h[1] = *(const v8h*)&Vt[p][j * 16 + lr][ks * 32 + lh * 8 + 16];
                o[j] = __builtin_amdgcn_wmma_f32_16x16x32_f16(
                    false, ua.v, false, ub.v, (short)0, o[j], false, false);
            }
        }
        __syncthreads();
    }

    // ---- epilogue: normalize and store f16
#pragma unroll
    for (int r = 0; r < 8; ++r) {
        float inv = 1.0f / l_run[r];
        int grow = q0 + wq_ * 16 + r + lh * 8;
#pragma unroll
        for (int j = 0; j < 4; ++j)
            oh[(size_t)grow * HHD + h * HD + j * 16 + lr] =
                (_Float16)(o[j][r] * inv);
    }
}

// ---------------------------------------------------------------------------
// Host launcher
// ---------------------------------------------------------------------------
extern "C" void kernel_launch(void* const* d_in, const int* in_sizes, int n_in,
                              void* d_out, int out_size, void* d_ws, size_t ws_size,
                              hipStream_t stream) {
    const float* x    = (const float*)d_in[0];
    const float* cosb = (const float*)d_in[1];
    const float* sinb = (const float*)d_in[2];
    const float* wq   = (const float*)d_in[3];
    const float* wk   = (const float*)d_in[4];
    const float* wv   = (const float*)d_in[5];
    const float* wo   = (const float*)d_in[6];
    float* out = (float*)d_out;

    char* ws = (char*)d_ws;
    size_t off = 0;
    auto carve = [&](size_t bytes) {
        void* ptr = ws + off;
        off += (bytes + 255) & ~(size_t)255;
        return ptr;
    };
    _Float16* xh   = (_Float16*)carve((size_t)S * D * 2);
    _Float16* wqT  = (_Float16*)carve((size_t)D * HHD * 2);     // (HHD, D)
    _Float16* wkT  = (_Float16*)carve((size_t)D * KVHD * 2);    // (KVHD, D)
    _Float16* wvT  = (_Float16*)carve((size_t)D * KVHD * 2);    // (KVHD, D)
    _Float16* woT  = (_Float16*)carve((size_t)HHD * D * 2);     // (D, HHD)
    _Float16* qhb  = (_Float16*)carve((size_t)S * HHD * 2);
    _Float16* khb  = (_Float16*)carve((size_t)S * KVHD * 2);
    _Float16* vtb  = (_Float16*)carve((size_t)S * KVHD * 2);    // (KVHD, S)
    _Float16* ohb  = (_Float16*)carve((size_t)S * HHD * 2);
    float*    f32scr = (float*)carve((size_t)S * HHD * 4);      // reused 3x

    // 1) downconvert: x plain; weights transposed (so GEMM B-tiles are
    //    contiguous row chunks -> direct async global->LDS copies)
    cvt_f16_kernel<<<(S * D + 255) / 256, 256, 0, stream>>>(x, xh, S * D);
    cvt_transpose_kernel<<<dim3(HHD / 32, D / 32), 256, 0, stream>>>(wq, wqT, D, HHD);
    cvt_transpose_kernel<<<dim3(KVHD / 32, D / 32), 256, 0, stream>>>(wk, wkT, D, KVHD);
    cvt_transpose_kernel<<<dim3(KVHD / 32, D / 32), 256, 0, stream>>>(wv, wvT, D, KVHD);
    cvt_transpose_kernel<<<dim3(D / 32, HHD / 32), 256, 0, stream>>>(wo, woT, HHD, D);

    // 2) Q = x@wq ; rope -> f16
    gemm_f16_kernel<<<dim3(S / BM, HHD / BN), 256, 0, stream>>>(
        xh, wqT, f32scr, S, HHD, D);
    rope_kernel<<<(S * H * 32 + 255) / 256, 256, 0, stream>>>(
        f32scr, cosb, sinb, qhb, H);

    // 3) K = x@wk ; rope -> f16
    gemm_f16_kernel<<<dim3(S / BM, KVHD / BN), 256, 0, stream>>>(
        xh, wkT, f32scr, S, KVHD, D);
    rope_kernel<<<(S * KV * 32 + 255) / 256, 256, 0, stream>>>(
        f32scr, cosb, sinb, khb, KV);

    // 4) V = x@wv ; convert+transpose -> vtb (KVHD, S)
    gemm_f16_kernel<<<dim3(S / BM, KVHD / BN), 256, 0, stream>>>(
        xh, wvT, f32scr, S, KVHD, D);
    cvt_transpose_kernel<<<dim3(KVHD / 32, S / 32), 256, 0, stream>>>(
        f32scr, vtb, S, KVHD);

    // 5) causal GQA flash attention
    attn_kernel<<<dim3(H, S / 64), 128, 0, stream>>>(qhb, khb, vtb, ohb);

    // 6) out = attn @ wo (f32 result)
    gemm_f16_kernel<<<dim3(S / BM, D / BN), 256, 0, stream>>>(
        ohb, woT, out, S, D, HHD);
}
